// ClusterPooling_54348516164012
// MI455X (gfx1250) — matrix operand order repaired and checked by reading, not compiled, not run on hardware
//
#include <hip/hip_runtime.h>
#include <math.h>

typedef __attribute__((ext_vector_type(16))) _Float16 v16h;
typedef __attribute__((ext_vector_type(8)))  _Float16 v8h;
typedef __attribute__((ext_vector_type(8)))  float    v8f;

#define TB 256

// ---------- small utility kernels ----------
__global__ void k_f32_to_f16(const float* __restrict__ x, _Float16* __restrict__ xh, long n) {
    long i = (long)blockIdx.x * blockDim.x + threadIdx.x;
    if (i < n) xh[i] = (_Float16)x[i];
}

__global__ void k_zero_f32(float* __restrict__ p, long n) {
    long i = (long)blockIdx.x * blockDim.x + threadIdx.x;
    if (i < n) p[i] = 0.0f;
}

__global__ void k_init_nodes(int* __restrict__ lab, int* __restrict__ deg,
                             int* __restrict__ bi, int n) {
    int i = blockIdx.x * blockDim.x + threadIdx.x;
    if (i < n) { lab[i] = i; deg[i] = 0; bi[i] = 0; }
}

// ---------- edge scores via WMMA ----------
// One wave computes scores for 16 edges.  A fragment: 16 gathered f16 feature
// rows (16x32 per step, C/32 steps for src*w1 then dst*w2).  B fragment: the
// weight chunk broadcast to all 16 columns, so every column of D equals the
// desired dot product.  Accumulate in f32, then tanh(acc + bias).
__global__ void k_edge_scores_wmma(const _Float16* __restrict__ xh,
                                   const _Float16* __restrict__ wh,
                                   const int* __restrict__ src,
                                   const int* __restrict__ dst,
                                   const float* __restrict__ bias,
                                   float* __restrict__ score,
                                   int E, int C) {
    const int lane  = threadIdx.x & 31;
    const int wid   = blockIdx.x * (blockDim.x >> 5) + (threadIdx.x >> 5);
    const int ebase = wid << 4;
    if (ebase >= E) return;                 // uniform per wave (E % 16 == 0)

    const int m  = lane & 15;               // matrix row handled by this lane
    const int hi = lane >> 4;               // half-wave selector
    const int se = src[ebase + m];
    const int de = dst[ebase + m];
    const _Float16* srow = xh + (long)se * C;
    const _Float16* drow = xh + (long)de * C;

    // Per-lane local-K slices per the 16-bit A / B VGPR layouts (ISA 7.12.2):
    // A: lanes 0-15 hold K{0..7,16..23}, lanes 16-31 hold K{8..15,24..31}
// B: lanes 0-15 hold K{0..7,16..23}, lanes 16-31 hold K{8..15,24..31}
    const int ak0 = hi ? 8  : 0;
    const int ak1 = hi ? 24 : 16;
    const int bk  = hi ? 16 : 0;

    v8f acc = {};
    #pragma unroll
    for (int sel = 0; sel < 2; ++sel) {
        const _Float16* row = sel ? drow : srow;
        const _Float16* wv  = wh + sel * C;
        for (int kc = 0; kc < C; kc += 32) {
            union { v16h v; v8h h[2]; } a, b;
            a.h[0] = *(const v8h*)(row + kc + ak0);
            a.h[1] = *(const v8h*)(row + kc + ak1);
            b.h[0] = *(const v8h*)(wv  + kc + ak0);
            b.h[1] = *(const v8h*)(wv  + kc + ak1);
            acc = __builtin_amdgcn_wmma_f32_16x16x32_f16(
                      false, a.v, false, b.v, (short)0, acc, false, false);
        }
    }

    // C/D layout: lane 0 holds D[0..7,0] in acc[0..7]; lane 16 holds D[8..15,0].
    const float bb = bias[0];
    if (lane == 0) {
        #pragma unroll
        for (int r = 0; r < 8; ++r) score[ebase + r] = tanhf(acc[r] + bb);
    } else if (lane == 16) {
        #pragma unroll
        for (int r = 0; r < 8; ++r) score[ebase + 8 + r] = tanhf(acc[r] + bb);
    }
}

// ---------- connected components (min-label + pointer jumping) ----------
__global__ void k_relax(const int* __restrict__ src, const int* __restrict__ dst,
                        const float* __restrict__ score, int* __restrict__ lab, int E) {
    int e = blockIdx.x * blockDim.x + threadIdx.x;
    if (e < E && score[e] > 0.0f) {
        int s = src[e], d = dst[e];
        int ls = lab[s], ld = lab[d];
        int mn = ls < ld ? ls : ld;
        atomicMin(&lab[s], mn);
        atomicMin(&lab[d], mn);
    }
}

__global__ void k_jump(const int* __restrict__ in, int* __restrict__ out, int n) {
    int i = blockIdx.x * blockDim.x + threadIdx.x;
    if (i < n) out[i] = in[in[i]];
}

// ---------- degree of contracted edges ----------
__global__ void k_deg(const int* __restrict__ src, const int* __restrict__ dst,
                      const float* __restrict__ score, int* __restrict__ deg, int E) {
    int e = blockIdx.x * blockDim.x + threadIdx.x;
    if (e < E && score[e] > 0.0f) {
        atomicAdd(&deg[src[e]], 1);
        atomicAdd(&deg[dst[e]], 1);
    }
}

// ---------- sx = scatter(src, score * x[dst]) : one wave per edge ----------
__global__ void k_scatter_sx(const int* __restrict__ src, const int* __restrict__ dst,
                             const float* __restrict__ score, const float* __restrict__ x,
                             float* __restrict__ sx, int E, int C) {
    int lane = threadIdx.x & 31;
    int e = blockIdx.x * (blockDim.x >> 5) + (threadIdx.x >> 5);
    if (e >= E) return;
    float s = score[e];
    const float* xr = x  + (long)dst[e] * C;
    float*       sr = sx + (long)src[e] * C;
    for (int c = lane * 4; c < C; c += 128) {
        float4 v = *(const float4*)(xr + c);
        atomicAdd(sr + c + 0, s * v.x);
        atomicAdd(sr + c + 1, s * v.y);
        atomicAdd(sr + c + 2, s * v.z);
        atomicAdd(sr + c + 3, s * v.w);
    }
}

// ---------- x_out = segment_sum(sx + single*x, cluster) ----------
__global__ void k_combine(const float* __restrict__ sx, const float* __restrict__ x,
                          const int* __restrict__ deg, const int* __restrict__ lab,
                          float* __restrict__ xout, int n, int C) {
    long i = (long)blockIdx.x * blockDim.x + threadIdx.x;
    if (i >= (long)n * C) return;
    int node = (int)(i / C);
    int c    = (int)(i - (long)node * C);
    float v = sx[i];
    if (deg[node] == 0) v += x[i];
    atomicAdd(&xout[(long)lab[node] * C + c], v);
}

// ---------- coarse edge index (as float, per tuple-concat d_out) ----------
__global__ void k_edge_out(const int* __restrict__ src, const int* __restrict__ dst,
                           const int* __restrict__ lab, float* __restrict__ eo, int E) {
    int e = blockIdx.x * blockDim.x + threadIdx.x;
    if (e < E) {
        eo[e]     = (float)lab[src[e]];
        eo[E + e] = (float)lab[dst[e]];
    }
}

// ---------- batch_out scatter (atomicMax for determinism) ----------
__global__ void k_batch_scatter(const int* __restrict__ lab, const int* __restrict__ batch,
                                int* __restrict__ bi, int n) {
    int i = blockIdx.x * blockDim.x + threadIdx.x;
    if (i < n) atomicMax(&bi[lab[i]], batch[i]);
}

__global__ void k_finalize(const int* __restrict__ bi, const int* __restrict__ lab,
                           float* __restrict__ bout, float* __restrict__ cout, int n) {
    int i = blockIdx.x * blockDim.x + threadIdx.x;
    if (i < n) { bout[i] = (float)bi[i]; cout[i] = (float)lab[i]; }
}

// ---------- host launcher ----------
extern "C" void kernel_launch(void* const* d_in, const int* in_sizes, int n_in,
                              void* d_out, int out_size, void* d_ws, size_t ws_size,
                              hipStream_t stream) {
    (void)n_in; (void)out_size; (void)ws_size;
    const float* x     = (const float*)d_in[0];
    const int*   ei    = (const int*)  d_in[1];
    const int*   batch = (const int*)  d_in[2];
    const float* w     = (const float*)d_in[3];
    const float* b     = (const float*)d_in[4];

    const int N = in_sizes[2];
    const int E = in_sizes[1] / 2;
    const int C = in_sizes[3] / 2;
    const long NC = (long)N * C;
    const int* src = ei;
    const int* dst = ei + E;

    // carve workspace (256B aligned slices)
    char* ws = (char*)d_ws;
    size_t off = 0;
    auto carve = [&](size_t bytes) -> void* {
        void* p = ws + off;
        off += (bytes + 255) & ~(size_t)255;
        return p;
    };
    _Float16* xh   = (_Float16*)carve((size_t)NC * 2);
    _Float16* wh   = (_Float16*)carve((size_t)2 * C * 2);
    float*    sc   = (float*)   carve((size_t)E * 4);
    int*      labA = (int*)     carve((size_t)N * 4);
    int*      labB = (int*)     carve((size_t)N * 4);
    int*      deg  = (int*)     carve((size_t)N * 4);
    int*      bi   = (int*)     carve((size_t)N * 4);
    float*    sx   = (float*)   carve((size_t)NC * 4);

    float* xout = (float*)d_out;            // [N*C]
    float* eout = xout + NC;                // [2*E]
    float* bout = eout + (size_t)2 * E;     // [N]
    float* cout = bout + N;                 // [N]

    const int nbNC = (int)((NC + TB - 1) / TB);
    const int nbN  = (N + TB - 1) / TB;
    const int nbE  = (E + TB - 1) / TB;

    // precision conversion + init
    k_f32_to_f16<<<nbNC, TB, 0, stream>>>(x, xh, NC);
    k_f32_to_f16<<<1, TB, 0, stream>>>(w, wh, 2 * C);
    k_zero_f32<<<nbNC, TB, 0, stream>>>(sx, NC);
    k_zero_f32<<<nbNC, TB, 0, stream>>>(xout, NC);
    k_init_nodes<<<nbN, TB, 0, stream>>>(labA, deg, bi, N);

    // edge scores: 1 wave per 16 edges, 8 waves per block
    {
        int waves  = (E + 15) / 16;
        int blocks = (waves + 7) / 8;
        k_edge_scores_wmma<<<blocks, TB, 0, stream>>>(xh, wh, src, dst, b, sc, E, C);
    }

    // connected components: ceil(log2(N)) + 4 iterations of relax + jump
    int iters = 4;
    for (long v = 1; v < N; v <<= 1) iters++;
    int* cur = labA;
    int* nxt = labB;
    for (int t = 0; t < iters; ++t) {
        k_relax<<<nbE, TB, 0, stream>>>(src, dst, sc, cur, E);
        k_jump<<<nbN, TB, 0, stream>>>(cur, nxt, N);
        int* tmp = cur; cur = nxt; nxt = tmp;
    }

    k_deg<<<nbE, TB, 0, stream>>>(src, dst, sc, deg, E);

    // sx scatter: one wave per edge (E*32 threads)
    {
        long threads = (long)E * 32;
        int blocks = (int)((threads + TB - 1) / TB);
        k_scatter_sx<<<blocks, TB, 0, stream>>>(src, dst, sc, x, sx, E, C);
    }

    k_combine<<<nbNC, TB, 0, stream>>>(sx, x, deg, cur, xout, N, C);
    k_edge_out<<<nbE, TB, 0, stream>>>(src, dst, cur, eout, E);
    k_batch_scatter<<<nbN, TB, 0, stream>>>(cur, batch, bi, N);
    k_finalize<<<nbN, TB, 0, stream>>>(bi, cur, bout, cout, N);
}
